// Model_9113920602479
// MI455X (gfx1250) — matrix operand order repaired and checked by reading, not compiled, not run on hardware
//
#include <hip/hip_runtime.h>

// ---------------------------------------------------------------------------
// CDNA5 (gfx1250) wave32 WMMA implementation of the conv-stack + MoE head.
// All heavy math runs through v_wmma_f32_16x16x32_bf16, with dual independent
// accumulator chains per tile to hide the bf16 WMMA->WMMA RAW hazard (4 coexec
// slots) behind the sibling chain's WMMA.
// ---------------------------------------------------------------------------

typedef __attribute__((ext_vector_type(16))) __bf16 v16bf;
typedef __attribute__((ext_vector_type(8)))  float  v8f;
typedef __attribute__((ext_vector_type(4)))  int    v4i;

// Packed-weight region layout inside d_ws (dword offsets, after 128 stat dwords)
#define OFF_C2   0        // conv2  B-frags: 9 ksteps x 2 ntiles x 256 dw
#define OFF_C3   4608     // conv3  B-frags
#define OFF_P1   9216     // pw1    B-frags: 9 x 8
#define OFF_P2   27648    // pw2    B-frags: 4 x 8
#define OFF_E1   35840    // ew1[3] B-frags: 3 x (9 x 8)
#define OFF_E2   91136    // ew2[3] B-frags: 3 x (4 x 8)
#define OFF_E3   115712   // ew3[3] B-frags: 3 x (4 x 1), N padded 10->16
#define PACK_TOT 118784
#define WS_PACK_DW 256            // pack region starts at byte 1024
#define WS_Y3_BYTE (512*1024)     // flattened conv output [B][288] f32

__device__ __forceinline__ unsigned short f2bf(float f) {
  unsigned int u = __builtin_bit_cast(unsigned int, f);
  unsigned int r = u + 0x7FFFu + ((u >> 16) & 1u);   // round-to-nearest-even
  return (unsigned short)(r >> 16);
}
__device__ __forceinline__ float bf2f(unsigned short h) {
  unsigned int u = ((unsigned int)h) << 16;
  return __builtin_bit_cast(float, u);
}

union FragU { v4i i2[2]; v16bf v; };

__device__ __forceinline__ v8f wmma_bf16(v16bf a, v16bf b, v8f c) {
  return __builtin_amdgcn_wmma_f32_16x16x32_bf16(false, a, false, b,
                                                 (short)0, c, false, false);
}

#define ZERO8 {0.f, 0.f, 0.f, 0.f, 0.f, 0.f, 0.f, 0.f}

// Dual 16x16 tile GEMM sharing the A fragment: two independent accumulator
// chains so consecutive WMMAs never have a RAW hazard.
__device__ __forceinline__ void gemm_tile2(const unsigned short* sA, int rowStrideB,
                                           int ksteps,
                                           const unsigned int* bf0,
                                           const unsigned int* bf1,
                                           int ksStrideDw, int lane,
                                           v8f& c0, v8f& c1) {
  c0 = (v8f)ZERO8;
  c1 = (v8f)ZERO8;
  const int sel = (lane >> 4) << 4;  // K-half byte select (0 or 16)
  const char* rb = (const char*)sA + (lane & 15) * rowStrideB;
  for (int ks = 0; ks < ksteps; ++ks) {
    FragU ua, ub0, ub1;
    const char* p = rb + ks * 64;
    ua.i2[0] = *(const v4i*)(p + sel);
    ua.i2[1] = *(const v4i*)(p + 32 + sel);
    const char* q0 = (const char*)(bf0 + ks * ksStrideDw) + lane * 32;
    const char* q1 = (const char*)(bf1 + ks * ksStrideDw) + lane * 32;
    ub0.i2[0] = *(const v4i*)(q0);
    ub0.i2[1] = *(const v4i*)(q0 + 16);
    ub1.i2[0] = *(const v4i*)(q1);
    ub1.i2[1] = *(const v4i*)(q1 + 16);
    c0 = wmma_bf16(ua.v, ub0.v, c0);
    c1 = wmma_bf16(ua.v, ub1.v, c1);
  }
}

// Single-tile GEMM with two interleaved K chains (for the 16-wide expert head)
__device__ __forceinline__ v8f gemm_tile_split(const unsigned short* sA, int rowStrideB,
                                               int ksteps, const unsigned int* bfrag,
                                               int ksStrideDw, int lane) {
  v8f ca = (v8f)ZERO8, cb = (v8f)ZERO8;
  const int sel = (lane >> 4) << 4;
  const char* rb = (const char*)sA + (lane & 15) * rowStrideB;
  for (int ks = 0; ks < ksteps; ks += 2) {
    FragU ua0, ua1, ub0, ub1;
    const char* p0 = rb + ks * 64;
    const char* p1 = rb + (ks + 1) * 64;
    ua0.i2[0] = *(const v4i*)(p0 + sel);
    ua0.i2[1] = *(const v4i*)(p0 + 32 + sel);
    ua1.i2[0] = *(const v4i*)(p1 + sel);
    ua1.i2[1] = *(const v4i*)(p1 + 32 + sel);
    const char* q0 = (const char*)(bfrag + ks * ksStrideDw) + lane * 32;
    const char* q1 = (const char*)(bfrag + (ks + 1) * ksStrideDw) + lane * 32;
    ub0.i2[0] = *(const v4i*)(q0);
    ub0.i2[1] = *(const v4i*)(q0 + 16);
    ub1.i2[0] = *(const v4i*)(q1);
    ub1.i2[1] = *(const v4i*)(q1 + 16);
    ca = wmma_bf16(ua0.v, ub0.v, ca);
    cb = wmma_bf16(ua1.v, ub1.v, cb);
  }
  return ca + cb;
}

// ---------------------------------------------------------------------------
// Kernel 0: zero BN stats + repack all GEMM weights into B-fragment layout.
// B-frag (K=32 x N=16): lane = ((k>>3)&1)*16 + n; dword v holds K {k0,k0+1}, k0 = 8*(lane>>4)+2v+8*(v>>2).
// conv K-order: k = (ky*3+kx)*32 + c  (channel-last im2col).
// ---------------------------------------------------------------------------
__global__ void pack_kernel(const float* __restrict__ cw2, const float* __restrict__ cw3,
                            const float* __restrict__ pw1, const float* __restrict__ pw2,
                            const float* __restrict__ ew1, const float* __restrict__ ew2,
                            const float* __restrict__ ew3, unsigned int* __restrict__ ws) {
  unsigned int* pk = ws + WS_PACK_DW;
  const int stride = gridDim.x * blockDim.x;
  for (int i = blockIdx.x * blockDim.x + threadIdx.x; i < 128 + PACK_TOT; i += stride) {
    if (i < 128) { ws[i] = 0u; continue; }   // sum/sumsq/scale/shift
    const int j = i - 128;
    float v0 = 0.f, v1 = 0.f;
    if (j < OFF_P1) {                                   // conv2 / conv3
      const float* w = (j < OFF_C3) ? cw2 : cw3;
      int t  = (j < OFF_C3) ? j : (j - OFF_C3);
      int fi = t >> 8, ln = (t >> 3) & 31, v = t & 7;
      int ks = fi >> 1, nt = fi & 1;
      int ky = ks / 3, kx = ks % 3;
      int n  = nt * 16 + (ln & 15);
      int c0 = ((ln >> 4) << 3) + 2 * v + ((v >> 2) << 3);
      v0 = w[((n * 32 + c0) * 3 + ky) * 3 + kx];
      v1 = w[((n * 32 + c0 + 1) * 3 + ky) * 3 + kx];
    } else if (j < OFF_E1) {                            // pw1 / pw2 (dense [K][128])
      const float* w; int t;
      if (j < OFF_P2) { w = pw1; t = j - OFF_P1; }
      else            { w = pw2; t = j - OFF_P2; }
      int fi = t >> 8, ln = (t >> 3) & 31, v = t & 7;
      int nt = fi & 7, ks = fi >> 3;
      int k0 = ks * 32 + ((ln >> 4) << 3) + 2 * v + ((v >> 2) << 3);
      int n  = nt * 16 + (ln & 15);
      v0 = w[k0 * 128 + n];
      v1 = w[(k0 + 1) * 128 + n];
    } else if (j < OFF_E3) {                            // ew1 / ew2 ([3][K][128])
      const float* w; int t, perE, Kdim;
      if (j < OFF_E2) { w = ew1; t = j - OFF_E1; perE = 72 * 256; Kdim = 288; }
      else            { w = ew2; t = j - OFF_E2; perE = 32 * 256; Kdim = 128; }
      int e = t / perE; int t2 = t - e * perE;
      int fi = t2 >> 8, ln = (t2 >> 3) & 31, v = t2 & 7;
      int nt = fi & 7, ks = fi >> 3;
      int k0 = ks * 32 + ((ln >> 4) << 3) + 2 * v + ((v >> 2) << 3);
      int n  = nt * 16 + (ln & 15);
      v0 = w[e * Kdim * 128 + k0 * 128 + n];
      v1 = w[e * Kdim * 128 + (k0 + 1) * 128 + n];
    } else {                                            // ew3 [3][128][10] -> N pad 16
      int t = j - OFF_E3;
      int e = t >> 10; int t2 = t & 1023;
      int ks = t2 >> 8, ln = (t2 >> 3) & 31, v = t2 & 7;
      int k0 = ks * 32 + ((ln >> 4) << 3) + 2 * v + ((v >> 2) << 3);
      int n  = ln & 15;
      if (n < 10) {
        v0 = ew3[e * 1280 + k0 * 10 + n];
        v1 = ew3[e * 1280 + (k0 + 1) * 10 + n];
      }
    }
    pk[j] = (unsigned int)f2bf(v0) | ((unsigned int)f2bf(v1) << 16);
  }
}

// ---------------------------------------------------------------------------
// Kernel 1: one block = one image. conv1 (VALU) -> conv2 (WMMA) -> conv3
// (WMMA), each fused with relu + 2x2 maxpool. Emits [288] f32 (flatten order
// c*9+s) + per-channel sum/sumsq atomics for BatchNorm.
// ---------------------------------------------------------------------------
__global__ __launch_bounds__(128)
void conv_stack_kernel(const float* __restrict__ x,
                       const float* __restrict__ cw1, const float* __restrict__ cb1,
                       const float* __restrict__ cb2, const float* __restrict__ cb3,
                       const unsigned int* __restrict__ pack,
                       float* __restrict__ y3, float* __restrict__ stats) {
  __shared__ __align__(16) float          s_in[30 * 30];        // conv1 input, pad 1
  __shared__ float                        s_w1[288];
  __shared__ float                        s_b1[32];
  __shared__ __align__(16) unsigned short s_t1[16 * 16 * 32];   // conv1 pooled, halo, [y][x][c] bf16
  __shared__ __align__(16) unsigned short s_c2[14 * 14 * 32];   // conv2 relu out, [p][c] bf16
  __shared__ __align__(16) unsigned short s_t2[9 * 9 * 32];     // conv2 pooled, halo, [y][x][c] bf16
  __shared__ float                        s_c3[49 * 32];        // conv3 relu out, [p][c] f32
  __shared__ float                        s_red[64];

  const int tid  = threadIdx.x;
  const int lane = tid & 31;
  const int wv   = tid >> 5;
  const int b    = blockIdx.x;

  for (int i = tid; i < 900; i += 128) {
    int yy = i / 30, xx = i % 30;
    float v = 0.f;
    if (yy >= 1 && yy <= 28 && xx >= 1 && xx <= 28)
      v = x[b * 784 + (yy - 1) * 28 + (xx - 1)];
    s_in[i] = v;
  }
  for (int i = tid; i < 288; i += 128) s_w1[i] = cw1[i];
  if (tid < 32) s_b1[tid] = cb1[tid];
  for (int i = tid; i < 16 * 16 * 32; i += 128) s_t1[i] = 0;
  for (int i = tid; i < 9 * 9 * 32;  i += 128) s_t2[i] = 0;
  if (tid < 64) s_red[tid] = 0.f;
  __syncthreads();

  // ---- conv1 + relu + pool (direct VALU; channel fixed per thread) ----
  {
    const int ch = tid & 31;
    float w[9];
#pragma unroll
    for (int t = 0; t < 9; ++t) w[t] = s_w1[ch * 9 + t];
    const float bb = s_b1[ch];
    for (int i = tid; i < 14 * 14 * 32; i += 128) {
      int p = i >> 5;
      int py = p / 14, px = p % 14;
      float m = -3.4e38f;
#pragma unroll
      for (int dy = 0; dy < 2; ++dy)
#pragma unroll
        for (int dx = 0; dx < 2; ++dx) {
          int oy = 2 * py + dy, ox = 2 * px + dx;
          float acc = 0.f;
#pragma unroll
          for (int ky = 0; ky < 3; ++ky)
#pragma unroll
            for (int kx = 0; kx < 3; ++kx)
              acc += s_in[(oy + ky) * 30 + ox + kx] * w[ky * 3 + kx];
          m = fmaxf(m, acc);
        }
      s_t1[((py + 1) * 16 + (px + 1)) * 32 + ch] = f2bf(fmaxf(m + bb, 0.f));
    }
  }
  __syncthreads();

  // ---- conv2: implicit GEMM, M=196 N=32 K=288; both N-tiles per M-tile ----
  for (int mt = wv; mt < 13; mt += 4) {
    int m = mt * 16 + (lane & 15);
    if (m > 195) m = 195;                 // garbage rows clamped (stores guarded)
    int oy = m / 14, ox = m % 14;
    const int sel = (lane >> 4) << 4;
    v8f c0 = (v8f)ZERO8, c1 = (v8f)ZERO8;
#pragma unroll
    for (int ks = 0; ks < 9; ++ks) {
      int ky = ks / 3, kx = ks % 3;
      const char* rb = (const char*)s_t1 + ((oy + ky) * 16 + (ox + kx)) * 64;
      FragU ua, ub0, ub1;
      ua.i2[0] = *(const v4i*)(rb + sel);
      ua.i2[1] = *(const v4i*)(rb + 32 + sel);
      const char* q0 = (const char*)(pack + OFF_C2 + (ks * 2) * 256) + lane * 32;
      const char* q1 = (const char*)(pack + OFF_C2 + (ks * 2 + 1) * 256) + lane * 32;
      ub0.i2[0] = *(const v4i*)(q0);
      ub0.i2[1] = *(const v4i*)(q0 + 16);
      ub1.i2[0] = *(const v4i*)(q1);
      ub1.i2[1] = *(const v4i*)(q1 + 16);
      c0 = wmma_bf16(ua.v, ub0.v, c0);
      c1 = wmma_bf16(ua.v, ub1.v, c1);
    }
    int ch0 = lane & 15;
    float b0 = cb2[ch0], b1 = cb2[16 + ch0];
#pragma unroll
    for (int r = 0; r < 8; ++r) {
      int mm = mt * 16 + r + ((lane >> 4) << 3);
      if (mm < 196) {
        s_c2[mm * 32 + ch0]      = f2bf(fmaxf(c0[r] + b0, 0.f));
        s_c2[mm * 32 + 16 + ch0] = f2bf(fmaxf(c1[r] + b1, 0.f));
      }
    }
  }
  __syncthreads();

  // ---- pool 14x14 -> 7x7 into haloed t2 ----
  for (int i = tid; i < 7 * 7 * 32; i += 128) {
    int ch = i & 31, p = i >> 5;
    int py = p / 7, px = p % 7;
    float m =          bf2f(s_c2[((2 * py) * 14 + 2 * px) * 32 + ch]);
    m = fmaxf(m, bf2f(s_c2[((2 * py) * 14 + 2 * px + 1) * 32 + ch]));
    m = fmaxf(m, bf2f(s_c2[((2 * py + 1) * 14 + 2 * px) * 32 + ch]));
    m = fmaxf(m, bf2f(s_c2[((2 * py + 1) * 14 + 2 * px + 1) * 32 + ch]));
    s_t2[((py + 1) * 9 + (px + 1)) * 32 + ch] = f2bf(m);
  }
  __syncthreads();

  // ---- conv3: implicit GEMM, M=49 N=32 K=288; one M-tile per wave ----
  {
    int mt = wv;                          // 4 M-tiles, 4 waves
    int m = mt * 16 + (lane & 15);
    if (m > 48) m = 48;
    int oy = m / 7, ox = m % 7;
    const int sel = (lane >> 4) << 4;
    v8f c0 = (v8f)ZERO8, c1 = (v8f)ZERO8;
#pragma unroll
    for (int ks = 0; ks < 9; ++ks) {
      int ky = ks / 3, kx = ks % 3;
      const char* rb = (const char*)s_t2 + ((oy + ky) * 9 + (ox + kx)) * 64;
      FragU ua, ub0, ub1;
      ua.i2[0] = *(const v4i*)(rb + sel);
      ua.i2[1] = *(const v4i*)(rb + 32 + sel);
      const char* q0 = (const char*)(pack + OFF_C3 + (ks * 2) * 256) + lane * 32;
      const char* q1 = (const char*)(pack + OFF_C3 + (ks * 2 + 1) * 256) + lane * 32;
      ub0.i2[0] = *(const v4i*)(q0);
      ub0.i2[1] = *(const v4i*)(q0 + 16);
      ub1.i2[0] = *(const v4i*)(q1);
      ub1.i2[1] = *(const v4i*)(q1 + 16);
      c0 = wmma_bf16(ua.v, ub0.v, c0);
      c1 = wmma_bf16(ua.v, ub1.v, c1);
    }
    int ch0 = lane & 15;
    float b0 = cb3[ch0], b1 = cb3[16 + ch0];
#pragma unroll
    for (int r = 0; r < 8; ++r) {
      int mm = mt * 16 + r + ((lane >> 4) << 3);
      if (mm < 49) {
        s_c3[mm * 32 + ch0]      = fmaxf(c0[r] + b0, 0.f);
        s_c3[mm * 32 + 16 + ch0] = fmaxf(c1[r] + b1, 0.f);
      }
    }
  }
  __syncthreads();

  // ---- pool 7x7 -> 3x3, write flatten order (c*9+s), BN partial stats ----
  for (int i = tid; i < 288; i += 128) {
    int ch = i & 31, p = i >> 5;
    int py = p / 3, px = p % 3;
    float m =          s_c3[((2 * py) * 7 + 2 * px) * 32 + ch];
    m = fmaxf(m, s_c3[((2 * py) * 7 + 2 * px + 1) * 32 + ch]);
    m = fmaxf(m, s_c3[((2 * py + 1) * 7 + 2 * px) * 32 + ch]);
    m = fmaxf(m, s_c3[((2 * py + 1) * 7 + 2 * px + 1) * 32 + ch]);
    y3[b * 288 + ch * 9 + p] = m;
    atomicAdd(&s_red[ch], m);
    atomicAdd(&s_red[32 + ch], m * m);
  }
  __syncthreads();
  if (tid < 64) atomicAdd(&stats[tid], s_red[tid]);
}

// ---------------------------------------------------------------------------
// Kernel 2: BN finalize (biased batch stats -> scale/shift per channel)
// ---------------------------------------------------------------------------
__global__ void bn_finalize_kernel(float* stats, const float* g, const float* bvec,
                                   float inv_n) {
  int c = threadIdx.x;
  if (c < 32) {
    float mean = stats[c] * inv_n;
    float var  = stats[32 + c] * inv_n - mean * mean;
    float sc   = g[c] * rsqrtf(var + 1e-5f);
    stats[64 + c] = sc;
    stats[96 + c] = bvec[c] - mean * sc;
  }
}

// ---------------------------------------------------------------------------
// Kernel 3: one wave = 16 examples. BN apply -> policy MLP (WMMA) -> argmax
// -> all 3 expert chains (WMMA) with per-row selection.
// ---------------------------------------------------------------------------
__global__ __launch_bounds__(32)
void head_kernel(const float* __restrict__ y3, const float* __restrict__ stats,
                 const unsigned int* __restrict__ pack,
                 const float* __restrict__ pb1, const float* __restrict__ pb2,
                 const float* __restrict__ pw3, const float* __restrict__ pb3,
                 const float* __restrict__ eb1, const float* __restrict__ eb2,
                 const float* __restrict__ eb3,
                 float* __restrict__ out, int* __restrict__ outact) {
  __shared__ __align__(16) unsigned short s_y[16 * 296];   // BN'd features, bf16
  __shared__ __align__(16) unsigned short s_h1[16 * 136];
  __shared__ __align__(16) unsigned short s_h2[16 * 136];
  __shared__ int s_act[16];

  const int lane = threadIdx.x;
  const int e0   = blockIdx.x * 16;
  const int col  = lane & 15;
  const int rsel = (lane >> 4) << 3;

  for (int i = lane; i < 16 * 288; i += 32) {
    int row = i / 288, k = i - row * 288;
    int c = k / 9;
    float v = y3[(e0 + row) * 288 + k] * stats[64 + c] + stats[96 + c];
    s_y[row * 296 + k] = f2bf(v);
  }
  for (int i = lane; i < 16 * 8; i += 32)
    s_y[(i >> 3) * 296 + 288 + (i & 7)] = 0;
  __syncthreads();

  // policy L1: 288 -> 128, relu (N-tile pairs, dual accumulators)
  for (int nt = 0; nt < 8; nt += 2) {
    v8f c0, c1;
    gemm_tile2(s_y, 592, 9, pack + OFF_P1 + nt * 256,
               pack + OFF_P1 + (nt + 1) * 256, 2048, lane, c0, c1);
    int ch = nt * 16 + col;
    float b0 = pb1[ch], b1 = pb1[ch + 16];
#pragma unroll
    for (int r = 0; r < 8; ++r) {
      s_h1[(r + rsel) * 136 + ch]      = f2bf(fmaxf(c0[r] + b0, 0.f));
      s_h1[(r + rsel) * 136 + ch + 16] = f2bf(fmaxf(c1[r] + b1, 0.f));
    }
  }
  __syncthreads();
  // policy L2: 128 -> 128, relu
  for (int nt = 0; nt < 8; nt += 2) {
    v8f c0, c1;
    gemm_tile2(s_h1, 272, 4, pack + OFF_P2 + nt * 256,
               pack + OFF_P2 + (nt + 1) * 256, 2048, lane, c0, c1);
    int ch = nt * 16 + col;
    float b0 = pb2[ch], b1 = pb2[ch + 16];
#pragma unroll
    for (int r = 0; r < 8; ++r) {
      s_h2[(r + rsel) * 136 + ch]      = f2bf(fmaxf(c0[r] + b0, 0.f));
      s_h2[(r + rsel) * 136 + ch + 16] = f2bf(fmaxf(c1[r] + b1, 0.f));
    }
  }
  __syncthreads();
  // logits 128 -> 3 + argmax (lowest-index tie break, matching jnp.argmax)
  if (lane < 16) {
    float l0 = pb3[0], l1 = pb3[1], l2 = pb3[2];
    for (int k = 0; k < 128; ++k) {
      float h = bf2f(s_h2[lane * 136 + k]);
      l0 += h * pw3[k * 3 + 0];
      l1 += h * pw3[k * 3 + 1];
      l2 += h * pw3[k * 3 + 2];
    }
    int a = 0; float best = l0;
    if (l1 > best) { best = l1; a = 1; }
    if (l2 > best) { a = 2; }
    s_act[lane] = a;
    outact[e0 + lane] = a;
  }
  __syncthreads();

  // expert chains: compute all 3, select per row (same routing throughout)
  v8f oacc = (v8f)ZERO8;
  for (int e = 0; e < 3; ++e) {
    for (int nt = 0; nt < 8; nt += 2) {
      v8f c0, c1;
      gemm_tile2(s_y, 592, 9, pack + OFF_E1 + (e * 72 + nt) * 256,
                 pack + OFF_E1 + (e * 72 + nt + 1) * 256, 2048, lane, c0, c1);
      int ch = nt * 16 + col;
      float b0 = eb1[e * 128 + ch], b1 = eb1[e * 128 + ch + 16];
#pragma unroll
      for (int r = 0; r < 8; ++r) {
        s_h1[(r + rsel) * 136 + ch]      = f2bf(c0[r] + b0);   // no relu
        s_h1[(r + rsel) * 136 + ch + 16] = f2bf(c1[r] + b1);
      }
    }
    __syncthreads();
    for (int nt = 0; nt < 8; nt += 2) {
      v8f c0, c1;
      gemm_tile2(s_h1, 272, 4, pack + OFF_E2 + (e * 32 + nt) * 256,
                 pack + OFF_E2 + (e * 32 + nt + 1) * 256, 2048, lane, c0, c1);
      int ch = nt * 16 + col;
      float b0 = eb2[e * 128 + ch], b1 = eb2[e * 128 + ch + 16];
#pragma unroll
      for (int r = 0; r < 8; ++r) {
        s_h2[(r + rsel) * 136 + ch]      = f2bf(c0[r] + b0);
        s_h2[(r + rsel) * 136 + ch + 16] = f2bf(c1[r] + b1);
      }
    }
    __syncthreads();
    {
      v8f c = gemm_tile_split(s_h2, 272, 4, pack + OFF_E3 + e * 1024, 256, lane);
      float bias = (col < 10) ? eb3[e * 10 + col] : 0.f;
#pragma unroll
      for (int r = 0; r < 8; ++r)
        if (s_act[r + rsel] == e) oacc[r] += c[r] + bias;
    }
    __syncthreads();
  }
  if (col < 10) {
#pragma unroll
    for (int r = 0; r < 8; ++r)
      out[(e0 + r + rsel) * 10 + col] = oacc[r];
  }
}

// ---------------------------------------------------------------------------
extern "C" void kernel_launch(void* const* d_in, const int* in_sizes, int n_in,
                              void* d_out, int out_size, void* d_ws, size_t ws_size,
                              hipStream_t stream) {
  const float* x   = (const float*)d_in[0];
  const float* cw1 = (const float*)d_in[1];
  const float* cb1 = (const float*)d_in[2];
  const float* cw2 = (const float*)d_in[3];
  const float* cb2 = (const float*)d_in[4];
  const float* cw3 = (const float*)d_in[5];
  const float* cb3 = (const float*)d_in[6];
  const float* bng = (const float*)d_in[7];
  const float* bnb = (const float*)d_in[8];
  const float* pw1 = (const float*)d_in[9];
  const float* pb1 = (const float*)d_in[10];
  const float* pw2 = (const float*)d_in[11];
  const float* pb2 = (const float*)d_in[12];
  const float* pw3 = (const float*)d_in[13];
  const float* pb3 = (const float*)d_in[14];
  const float* ew1 = (const float*)d_in[15];
  const float* eb1 = (const float*)d_in[16];
  const float* ew2 = (const float*)d_in[17];
  const float* eb2 = (const float*)d_in[18];
  const float* ew3 = (const float*)d_in[19];
  const float* eb3 = (const float*)d_in[20];

  const int B = in_sizes[0] / 784;          // 4096

  unsigned int* ws    = (unsigned int*)d_ws;
  float*        stats = (float*)d_ws;                       // [0:32)=sum [32:64)=sumsq [64:96)=scale [96:128)=shift
  float*        y3    = (float*)((char*)d_ws + WS_Y3_BYTE); // [B][288]
  const unsigned int* pack = ws + WS_PACK_DW;

  pack_kernel<<<465, 256, 0, stream>>>(cw2, cw3, pw1, pw2, ew1, ew2, ew3, ws);
  conv_stack_kernel<<<B, 128, 0, stream>>>(x, cw1, cb1, cb2, cb3, pack, y3, stats);
  bn_finalize_kernel<<<1, 32, 0, stream>>>(stats, bng, bnb, 1.0f / (float(B) * 9.0f));
  head_kernel<<<B / 16, 32, 0, stream>>>(y3, stats, pack, pb1, pb2, pw3, pb3,
                                         eb1, eb2, eb3,
                                         (float*)d_out, (int*)d_out + B * 10);
}
